// ApagadoAleatorio_7567732376342
// MI455X (gfx1250) — matrix operand order repaired and checked by reading, not compiled, not run on hardware
//
#include <hip/hip_runtime.h>
#include <math.h>

typedef _Float16 half_t;
typedef __attribute__((ext_vector_type(8)))  _Float16 v8h;
typedef __attribute__((ext_vector_type(16))) _Float16 v16h;
typedef __attribute__((ext_vector_type(8)))  float    v8f;

#define DD 32      // d
#define HH 64      // h
#define XPAD 72    // row stride (halves) for 64-wide tiles: 144B, 16B-aligned
#define GPAD 40    // row stride (halves) for 32-wide tiles:  80B, 16B-aligned
#define TPW 4      // edge tiles per wave in msg kernel

__device__ __forceinline__ v16h cat16(v8h lo, v8h hi) {
    return __builtin_shufflevector(lo, hi, 0,1,2,3,4,5,6,7,8,9,10,11,12,13,14,15);
}
// Per-wave LDS fence: DS ops are in-order per wave; dscnt==0 + compiler barrier
// makes cross-lane LDS data visible without a block-wide s_barrier.
__device__ __forceinline__ void lds_fence() {
    asm volatile("s_wait_dscnt 0x0" ::: "memory");
}
__device__ __forceinline__ void atomic_fadd(float* p, float v) {
    __hip_atomic_fetch_add(p, v, __ATOMIC_RELAXED, __HIP_MEMORY_SCOPE_AGENT);
}

// ---------------------------------------------------------------------------
// Kernel A: edge message MLP + scatter-add.
//   msg = relu([ls[first]|ls[second]] @ Wm1 + bm1) @ Wm2 + bm2 ; ei[second] += msg
// 8 waves/block; each wave owns TPW 16-edge tiles with weight fragments
// resident in registers (12 x v16h).
// ---------------------------------------------------------------------------
__global__ __launch_bounds__(256) void msg_kernel(
    const float* __restrict__ ls,
    const int*   __restrict__ first,
    const int*   __restrict__ second,
    const float* __restrict__ Wm1, const float* __restrict__ bm1,
    const float* __restrict__ Wm2, const float* __restrict__ bm2,
    float* __restrict__ ei, int M)
{
    // fragment-ready packed weights: [frag][lane][16 halves], 32B/lane contiguous
    __shared__ __align__(32) half_t sW1f[8 * 32 * 16];   // 8 frags (kc*4+nc)
    __shared__ __align__(32) half_t sW2f[4 * 32 * 16];   // 4 frags (kc*2+nc)
    __shared__ __align__(32) half_t sX[8][16 * XPAD];
    __shared__ __align__(32) half_t sH[8][16 * XPAD];

    const int tid = threadIdx.x;
    // pack Wm1 (64x64): element (k,n) -> frag kc*4+nc, lane ((kt>>3)&1)*16+nt, half (kt>>4)*8+(kt&7)
    for (int i = tid; i < HH * HH; i += 256) {
        const int k = i >> 6, n = i & 63;
        const int kt = k & 31, nt = n & 15;
        const int f = (k >> 5) * 4 + (n >> 4);
        const int ln = (((kt >> 3) & 1) << 4) + nt;
        sW1f[((f * 32 + ln) << 4) + ((kt >> 4) << 3) + (kt & 7)] = (half_t)Wm1[i];
    }
    // pack Wm2 (64x32)
    for (int i = tid; i < HH * DD; i += 256) {
        const int k = i >> 5, n = i & 31;
        const int kt = k & 31, nt = n & 15;
        const int f = (k >> 5) * 2 + (n >> 4);
        const int ln = (((kt >> 3) & 1) << 4) + nt;
        sW2f[((f * 32 + ln) << 4) + ((kt >> 4) << 3) + (kt & 7)] = (half_t)Wm2[i];
    }
    __syncthreads();

    const int wave = tid >> 5, lane = tid & 31;
    const int g = lane >> 4, col = lane & 15, row = lane & 15;

    // resident weight fragments
    v16h w1f[8], w2f[4];
    #pragma unroll
    for (int f = 0; f < 8; ++f) w1f[f] = *(const v16h*)&sW1f[(f * 32 + lane) << 4];
    #pragma unroll
    for (int f = 0; f < 4; ++f) w2f[f] = *(const v16h*)&sW2f[(f * 32 + lane) << 4];

    const float b1 = bm1[col];          // per-lane biases for each nc handled below
    half_t* const xt = sX[wave];
    half_t* const ht = sH[wave];

    for (int t = 0; t < TPW; ++t) {
        const int base = ((blockIdx.x * 8 + wave) * TPW + t) * 16;

        // ---- gather X tile (16x64 f16): lane -> edge lane/2, half lane&1 ----
        {
            const int r = lane >> 1, hsel = lane & 1;
            int e = base + r; if (e >= M) e = M - 1;
            const int src = hsel ? second[e] : first[e];
            const float4* __restrict__ row4 = (const float4*)(ls + (long)src * DD);
            v8h* dst = (v8h*)&xt[r * XPAD + hsel * 32];
            #pragma unroll
            for (int c = 0; c < 4; ++c) {
                const float4 f0 = row4[2 * c], f1 = row4[2 * c + 1];
                v8h o;
                o[0]=(half_t)f0.x; o[1]=(half_t)f0.y; o[2]=(half_t)f0.z; o[3]=(half_t)f0.w;
                o[4]=(half_t)f1.x; o[5]=(half_t)f1.y; o[6]=(half_t)f1.z; o[7]=(half_t)f1.w;
                dst[c] = o;
            }
        }
        lds_fence();

        // ---- A fragments for GEMM1 (two K-chunks), vectorized 16B LDS loads ----
        v16h a0 = cat16(*(const v8h*)&xt[row * XPAD +  8 * g],
                        *(const v8h*)&xt[row * XPAD + 16 + 8 * g]);
        v16h a1 = cat16(*(const v8h*)&xt[row * XPAD + 32 + 8 * g],
                        *(const v8h*)&xt[row * XPAD + 48 + 8 * g]);

        // ---- GEMM1: H = relu(X @ Wm1 + bm1) ----
        #pragma unroll
        for (int nc = 0; nc < 4; ++nc) {
            v8f c = {};
            c = __builtin_amdgcn_wmma_f32_16x16x32_f16(false, a0, false, w1f[nc],
                                                       (short)0, c, false, false);
            c = __builtin_amdgcn_wmma_f32_16x16x32_f16(false, a1, false, w1f[4 + nc],
                                                       (short)0, c, false, false);
            const float bias = bm1[nc * 16 + col];
            #pragma unroll
            for (int r = 0; r < 8; ++r) {
                float v = c[r] + bias;
                v = v > 0.f ? v : 0.f;
                ht[(r + 8 * g) * XPAD + nc * 16 + col] = (half_t)v;
            }
        }
        lds_fence();

        // ---- A fragments for GEMM2 ----
        v16h a2 = cat16(*(const v8h*)&ht[row * XPAD +  8 * g],
                        *(const v8h*)&ht[row * XPAD + 16 + 8 * g]);
        v16h a3 = cat16(*(const v8h*)&ht[row * XPAD + 32 + 8 * g],
                        *(const v8h*)&ht[row * XPAD + 48 + 8 * g]);

        // ---- GEMM2: msg = H @ Wm2 + bm2, scatter-add ----
        #pragma unroll
        for (int nc = 0; nc < 2; ++nc) {
            v8f c = {};
            c = __builtin_amdgcn_wmma_f32_16x16x32_f16(false, a2, false, w2f[nc],
                                                       (short)0, c, false, false);
            c = __builtin_amdgcn_wmma_f32_16x16x32_f16(false, a3, false, w2f[2 + nc],
                                                       (short)0, c, false, false);
            const float bias = bm2[nc * 16 + col];
            #pragma unroll
            for (int r = 0; r < 8; ++r) {
                const int e = base + r + 8 * g;
                if (e < M) {
                    const int tgt = second[e];
                    atomic_fadd(&ei[(long)tgt * DD + nc * 16 + col], c[r] + bias);
                }
            }
        }
        lds_fence();   // protect sX/sH from next iteration's overwrite (WAR)
    }
    (void)b1;
}

// ---------------------------------------------------------------------------
// Kernel B: GRU update (Keras reset_after=True), in-place on ls.
// One wave = 16 nodes; K=32 -> single WMMA K-chunk; 6 N-tiles per matmul.
// ---------------------------------------------------------------------------
__global__ __launch_bounds__(256) void gru_kernel(
    float* __restrict__ ls, const float* __restrict__ ei,
    const float* __restrict__ Wx, const float* __restrict__ Wh,
    const float* __restrict__ bg, int N)
{
    __shared__ __align__(32) half_t sWf[12 * 32 * 16];   // frags: 0-5 Wx, 6-11 Wh
    __shared__ __align__(32) half_t sXt[8][16 * GPAD];
    __shared__ __align__(32) half_t sHt[8][16 * GPAD];

    const int tid = threadIdx.x;
    for (int i = tid; i < DD * 96; i += 256) {
        const int k = i / 96, n = i % 96;
        const int f = n >> 4;
        const int ln = (((k >> 3) & 1) << 4) + (n & 15);
        const int off = ((ln << 4) + ((k >> 4) << 3) + (k & 7));
        sWf[(f * 32) * 16 + off]       = (half_t)Wx[i];
        sWf[((f + 6) * 32) * 16 + off] = (half_t)Wh[i];
    }
    __syncthreads();

    const int wave = tid >> 5, lane = tid & 31;
    const int g = lane >> 4, col = lane & 15, row = lane & 15;
    const int base = (blockIdx.x * 8 + wave) * 16;
    half_t* const xt = sXt[wave];
    half_t* const ht = sHt[wave];

    {
        const int r = lane >> 1, hsel = lane & 1;
        int n = base + r; if (n >= N) n = N - 1;
        const float4* __restrict__ x4 = (const float4*)(ei + (long)n * DD + hsel * 16);
        const float4* __restrict__ h4 = (const float4*)(ls + (long)n * DD + hsel * 16);
        v8h* dx = (v8h*)&xt[r * GPAD + hsel * 16];
        v8h* dh = (v8h*)&ht[r * GPAD + hsel * 16];
        #pragma unroll
        for (int c = 0; c < 2; ++c) {
            const float4 f0 = x4[2 * c], f1 = x4[2 * c + 1];
            const float4 g0 = h4[2 * c], g1 = h4[2 * c + 1];
            v8h ox, oh;
            ox[0]=(half_t)f0.x; ox[1]=(half_t)f0.y; ox[2]=(half_t)f0.z; ox[3]=(half_t)f0.w;
            ox[4]=(half_t)f1.x; ox[5]=(half_t)f1.y; ox[6]=(half_t)f1.z; ox[7]=(half_t)f1.w;
            oh[0]=(half_t)g0.x; oh[1]=(half_t)g0.y; oh[2]=(half_t)g0.z; oh[3]=(half_t)g0.w;
            oh[4]=(half_t)g1.x; oh[5]=(half_t)g1.y; oh[6]=(half_t)g1.z; oh[7]=(half_t)g1.w;
            dx[c] = ox; dh[c] = oh;
        }
    }
    lds_fence();

    const v16h ax = cat16(*(const v8h*)&xt[row * GPAD + 8 * g],
                          *(const v8h*)&xt[row * GPAD + 16 + 8 * g]);
    const v16h ah = cat16(*(const v8h*)&ht[row * GPAD + 8 * g],
                          *(const v8h*)&ht[row * GPAD + 16 + 8 * g]);

    v8f gi[6], gh[6];
    #pragma unroll
    for (int nc = 0; nc < 6; ++nc) {
        const v16h bx = *(const v16h*)&sWf[(nc * 32 + lane) << 4];
        const v16h bh = *(const v16h*)&sWf[((nc + 6) * 32 + lane) << 4];
        v8f cz = {};
        gi[nc] = __builtin_amdgcn_wmma_f32_16x16x32_f16(false, ax, false, bx,
                                                        (short)0, cz, false, false);
        v8f ch = {};
        gh[nc] = __builtin_amdgcn_wmma_f32_16x16x32_f16(false, ah, false, bh,
                                                        (short)0, ch, false, false);
    }

    #pragma unroll
    for (int jc = 0; jc < 2; ++jc) {
        const int j = jc * 16 + col;
        const float bi_z = bg[j],      bh_z = bg[96 + j];
        const float bi_r = bg[32 + j], bh_r = bg[96 + 32 + j];
        const float bi_h = bg[64 + j], bh_h = bg[96 + 64 + j];
        #pragma unroll
        for (int r = 0; r < 8; ++r) {
            const int n = base + r + 8 * g;
            if (n < N) {
                const float z  = 1.f / (1.f + expf(-((gi[jc][r]     + bi_z) + (gh[jc][r]     + bh_z))));
                const float rr = 1.f / (1.f + expf(-((gi[2 + jc][r] + bi_r) + (gh[2 + jc][r] + bh_r))));
                const float hhat = tanhf((gi[4 + jc][r] + bi_h) + rr * (gh[4 + jc][r] + bh_h));
                const float hp = ls[(long)n * DD + j];
                ls[(long)n * DD + j] = z * hp + (1.f - z) * hhat;
            }
        }
    }
}

// ---------------------------------------------------------------------------
// Kernel C: graph segment-sum  gemb[gid[n]] += ls[n]
// ---------------------------------------------------------------------------
__global__ void seg_kernel(const float* __restrict__ ls,
                           const int* __restrict__ gids,
                           float* __restrict__ gemb, int N)
{
    const long t = (long)blockIdx.x * blockDim.x + threadIdx.x;
    if (t < (long)N * DD) {
        const int n = (int)(t >> 5), f = (int)(t & 31);
        atomic_fadd(&gemb[(long)gids[n] * DD + f], ls[t]);
    }
}

// ---------------------------------------------------------------------------
// Readout MLP (tiny): 64x32 -> 256 -> 256 -> 1
// ---------------------------------------------------------------------------
__global__ void ro1_kernel(const float* __restrict__ gemb,
                           const float* __restrict__ W, const float* __restrict__ b,
                           const float* __restrict__ mask, float* __restrict__ out)
{
    const int gidx = blockIdx.x, c = threadIdx.x;
    float acc = b[c];
    #pragma unroll
    for (int k = 0; k < 32; ++k) acc += gemb[gidx * 32 + k] * W[k * 256 + c];
    acc = acc > 0.f ? acc : 0.f;
    out[gidx * 256 + c] = acc * mask[c];
}

__global__ void ro2_kernel(const float* __restrict__ r1,
                           const float* __restrict__ W, const float* __restrict__ b,
                           const float* __restrict__ mask, float* __restrict__ out)
{
    const int gidx = blockIdx.x, c = threadIdx.x;
    float acc = b[c];
    for (int k = 0; k < 256; ++k) acc += r1[gidx * 256 + k] * W[k * 256 + c];
    acc = acc > 0.f ? acc : 0.f;
    out[gidx * 256 + c] = acc * mask[c];
}

__global__ void ro3_kernel(const float* __restrict__ r2,
                           const float* __restrict__ W, const float* __restrict__ b,
                           float* __restrict__ out)
{
    const int gidx = threadIdx.x;
    float acc = b[0];
    for (int k = 0; k < 256; ++k) acc += r2[gidx * 256 + k] * W[k];
    out[gidx] = acc;
}

// ---------------------------------------------------------------------------
extern "C" void kernel_launch(void* const* d_in, const int* in_sizes, int n_in,
                              void* d_out, int out_size, void* d_ws, size_t ws_size,
                              hipStream_t stream)
{
    const float* ls_in = (const float*)d_in[0];
    const int*   first  = (const int*)d_in[1];
    const int*   second = (const int*)d_in[2];
    const int*   gids   = (const int*)d_in[3];
    const float* Wm1 = (const float*)d_in[5];
    const float* bm1 = (const float*)d_in[6];
    const float* Wm2 = (const float*)d_in[7];
    const float* bm2 = (const float*)d_in[8];
    const float* Wx  = (const float*)d_in[9];
    const float* Wh  = (const float*)d_in[10];
    const float* bg  = (const float*)d_in[11];
    const float* Wr1 = (const float*)d_in[12];
    const float* br1 = (const float*)d_in[13];
    const float* mk1 = (const float*)d_in[14];
    const float* Wr2 = (const float*)d_in[15];
    const float* br2 = (const float*)d_in[16];
    const float* mk2 = (const float*)d_in[17];
    const float* Wr3 = (const float*)d_in[18];
    const float* br3 = (const float*)d_in[19];

    const int N = in_sizes[0] / DD;
    const int M = in_sizes[1];
    const long N32 = (long)N * DD;

    float* ws   = (float*)d_ws;
    float* ls   = ws;
    float* ei   = ls + N32;
    float* gemb = ei + N32;
    float* r1   = gemb + 64 * 32;
    float* r2   = r1 + 64 * 256;

    hipMemcpyAsync(ls, ls_in, N32 * sizeof(float), hipMemcpyDeviceToDevice, stream);

    const int edgesPerBlk = 8 * TPW * 16;          // 512
    const int blkA = (M + edgesPerBlk - 1) / edgesPerBlk;
    const int blkB = (N + 127) / 128;

    for (int t = 0; t < 4; ++t) {
        hipMemsetAsync(ei, 0, N32 * sizeof(float), stream);
        msg_kernel<<<blkA, 256, 0, stream>>>(ls, first, second, Wm1, bm1, Wm2, bm2, ei, M);
        gru_kernel<<<blkB, 256, 0, stream>>>(ls, ei, Wx, Wh, bg, N);
    }

    hipMemsetAsync(gemb, 0, 64 * 32 * sizeof(float), stream);
    const int blkC = (int)((N32 + 255) / 256);
    seg_kernel<<<blkC, 256, 0, stream>>>(ls, gids, gemb, N);

    ro1_kernel<<<64, 256, 0, stream>>>(gemb, Wr1, br1, mk1, r1);
    ro2_kernel<<<64, 256, 0, stream>>>(r1, Wr2, br2, mk2, r2);
    ro3_kernel<<<1, 64, 0, stream>>>(r2, Wr3, br3, (float*)d_out);
}
